// CustomRNN_68418829025465
// MI455X (gfx1250) — matrix operand order repaired and checked by reading, not compiled, not run on hardware
//
#include <hip/hip_runtime.h>
#include <hip/hip_bf16.h>

// ---------------------------------------------------------------------------
// 2-layer LSTM, B=64 T=256 N=512 H=1024, bf16 WMMA (f32 accumulate) on gfx1250.
//  * Weights packed once into the wave32 WMMA B-operand layout (bf16).
//  * Per (t, layer) fused kernel:
//      - A-rows ([x_t | h_prev], 16 rows per block) staged into LDS with
//        global_load_async_to_lds_b128, double-buffered in 512-K slabs,
//        synchronized with s_wait_asynccnt (ASYNCcnt pipeline).
//      - 16 v_wmma_f32_16x16x32_bf16 per slab per wave on TWO independent
//        accumulator chains (even/odd K) to cover XDL pipeline latency;
//        4 waves compute the i/f/g/o gate tiles of the same (b-,h-tile).
//      - Gate exchange through LDS, then LSTM cell update; h_new written as
//        fp32 (sequence out / h_final) and bf16 (next step's A operand,
//        double-buffered); c updated in place (one owner per element).
// ---------------------------------------------------------------------------

typedef __bf16 bf16_t;
typedef __attribute__((ext_vector_type(16))) __bf16 v16bf;
typedef __attribute__((ext_vector_type(8)))  float  v8f;

#define LSTM_H 1024
#define LDS_ROW 520            // 512 K-elements + 8 pad (bank-conflict-free)

// ---------------------------------------------------------------------------
// Pack Wcat (4H x Ktot fp32, Wcat[j][k] = k<K0 ? Wi[j][k] : Wh[j][k-K0]) into
// bf16 WMMA B-operand tiles: layout [jt (4H/16)][kc (Ktot/32)][lane 32][e 16].
// Lane n<16 holds column n, elements e = B[k0+e][n];
// lane n>=16 holds column n-16, elements e = B[k0+16+e][n-16].
// ---------------------------------------------------------------------------
__global__ void pack_weights_kernel(bf16_t* __restrict__ Wp,
                                    const float* __restrict__ Wi,
                                    const float* __restrict__ Wh,
                                    int K0, int Ktot, int total) {
  int p = blockIdx.x * blockDim.x + threadIdx.x;
  if (p >= total) return;
  int KC   = Ktot >> 5;
  int e    = p & 15;
  int lane = (p >> 4) & 31;
  int kc   = (p >> 9) % KC;
  int jt   = (p >> 9) / KC;
  int j = jt * 16 + (lane & 15);
  int k = kc * 32 + ((lane >> 4) << 4) + e;
  float v = (k < K0) ? Wi[(size_t)j * K0 + k]
                     : Wh[(size_t)j * LSTM_H + (k - K0)];
  Wp[p] = (bf16_t)v;
}

__global__ void bias_combine_kernel(float* __restrict__ bias,
                                    const float* __restrict__ bi,
                                    const float* __restrict__ bh, int n) {
  int i = blockIdx.x * blockDim.x + threadIdx.x;
  if (i < n) bias[i] = bi[i] + bh[i];
}

__global__ void f32_to_bf16_kernel(bf16_t* __restrict__ dst,
                                   const float* __restrict__ src, int n) {
  int i = blockIdx.x * blockDim.x + threadIdx.x;
  if (i < n) dst[i] = (bf16_t)src[i];
}

// h_in [2][B*H] -> bf16 state buffers; c_in [2][B*H] -> c region of d_out.
__global__ void init_state_kernel(bf16_t* __restrict__ h0bf,
                                  bf16_t* __restrict__ h1bf,
                                  float* __restrict__ cdst,
                                  const float* __restrict__ h_in,
                                  const float* __restrict__ c_in, int BH) {
  int i = blockIdx.x * blockDim.x + threadIdx.x;
  if (i >= 2 * BH) return;
  cdst[i] = c_in[i];
  bf16_t hv = (bf16_t)h_in[i];
  if (i < BH) h0bf[i] = hv;
  else        h1bf[i - BH] = hv;
}

// ---------------------------------------------------------------------------
// Async-stage one 512-K slab of the block's 16 A-rows into LDS buffer `buf`.
// 1024 16-byte chunks, 128 threads -> 8 async instructions per wave.
// ---------------------------------------------------------------------------
__device__ __forceinline__ void stage_to_lds(uint32_t sbase, int buf,
                                             const bf16_t* __restrict__ seg0,
                                             int seg0_stride, int K0,
                                             const bf16_t* __restrict__ seg1,
                                             int kbase, int tid) {
  const int H = LSTM_H;
#pragma unroll
  for (int i = 0; i < 8; ++i) {
    int q   = tid + i * 128;          // 0..1023
    int row = q >> 6;                 // 0..15
    int ke  = (q & 63) << 3;          // 0..504, element offset in slab
    int kg  = kbase + ke;             // global K index
    const bf16_t* src = (kg < K0)
        ? (seg0 + (size_t)row * seg0_stride + kg)
        : (seg1 + (size_t)row * H + (kg - K0));
    uint32_t dst = sbase + (uint32_t)(((buf * 16 + row) * LDS_ROW + ke) * 2);
    asm volatile("global_load_async_to_lds_b128 %0, %1, off"
                 :: "v"(dst), "v"(src) : "memory");
  }
}

// ---------------------------------------------------------------------------
// Fused LSTM step for one layer.
//   grid  = (H/16, B/16) -> (64, 4); block = (32, 4): wave g computes gate g.
// ---------------------------------------------------------------------------
__global__ __launch_bounds__(128)
void lstm_step_kernel(const bf16_t* __restrict__ seg0, int seg0_stride, int K0,
                      const bf16_t* __restrict__ seg1,   // h_prev [B][H]
                      const bf16_t* __restrict__ Wp,     // packed weights
                      const float*  __restrict__ bias,   // [4H]
                      float* __restrict__ cbuf,          // [B][H] in/out
                      float* __restrict__ hfin,          // [B][H] fp32 h out
                      float* __restrict__ outseq,        // optional seq out
                      int out_stride,
                      bf16_t* __restrict__ hnext) {      // [B][H] bf16 h out
  const int H    = LSTM_H;
  const int Ktot = K0 + H;
  const int KC   = Ktot >> 5;        // 32-wide K chunks
  const int S    = KC >> 4;          // 512-wide K slabs (3 or 4)

  const int lane = threadIdx.x;      // 0..31
  const int g    = threadIdx.y;      // gate 0..3 (i,f,g,o)
  const int jt   = blockIdx.x;       // 0..H/16-1
  const int bt   = blockIdx.y;       // 0..B/16-1
  const int tid  = g * 32 + lane;

  __shared__ bf16_t smemA[2 * 16 * LDS_ROW];
  __shared__ float  gx[4][8][32];
  const uint32_t sbase = (uint32_t)(uintptr_t)(void*)smemA;

  // Two independent accumulator chains (even/odd K chunks); bias in chain 0.
  const int col  = jt * 16 + (lane & 15);
  const float bv = bias[g * H + col];
  v8f acc0 = {bv, bv, bv, bv, bv, bv, bv, bv};
  v8f acc1 = {0.f, 0.f, 0.f, 0.f, 0.f, 0.f, 0.f, 0.f};

  // A-operand half-wave K offset (contiguous: lane half h holds k = 16h..16h+15).
  const int hsel = (lane >> 4) << 4;

  // B-operand: packed [jt_global][kc][lane][16]; 32B contiguous per lane.
  const int jt_global = g * (H / 16) + jt;
  const bf16_t* wptr = Wp + (size_t)jt_global * KC * 512 + (size_t)lane * 16;

  union AV { v16bf v; uint4 q[2]; };

  // Prime the async pipeline with slab 0.
  stage_to_lds(sbase, 0, seg0, seg0_stride, K0, seg1, 0, tid);

  for (int s = 0; s < S; ++s) {
    if (s + 1 < S)
      stage_to_lds(sbase, (s + 1) & 1, seg0, seg0_stride, K0, seg1,
                   (s + 1) << 9, tid);
    // Wait for slab s only; slab s+1 (8 async ops) may remain outstanding.
    if (s + 1 < S) asm volatile("s_wait_asynccnt 0x8" ::: "memory");
    else           asm volatile("s_wait_asynccnt 0x0" ::: "memory");
    __syncthreads();

    const bf16_t* arow = smemA + (((s & 1) * 16 + (lane & 15)) * LDS_ROW);
#pragma unroll
    for (int k2 = 0; k2 < 16; k2 += 2) {
      AV a0, b0, a1, b1;
      a0.q[0] = *(const uint4*)(arow + k2 * 32 + hsel);
      a0.q[1] = *(const uint4*)(arow + k2 * 32 + 8 + hsel);
      b0.q[0] = *(const uint4*)(wptr);
      b0.q[1] = *(const uint4*)(wptr + 8);
      a1.q[0] = *(const uint4*)(arow + (k2 + 1) * 32 + hsel);
      a1.q[1] = *(const uint4*)(arow + (k2 + 1) * 32 + 8 + hsel);
      b1.q[0] = *(const uint4*)(wptr + 512);
      b1.q[1] = *(const uint4*)(wptr + 520);
      acc0 = __builtin_amdgcn_wmma_f32_16x16x32_bf16(
          false, a0.v, false, b0.v, (short)0, acc0, false, false);
      acc1 = __builtin_amdgcn_wmma_f32_16x16x32_bf16(
          false, a1.v, false, b1.v, (short)0, acc1, false, false);
      wptr += 1024;
    }
    __syncthreads();   // everyone done reading this buffer before re-staging
  }

  // Merge chains and exchange the four gate tiles through LDS.
  // C/D layout: VGPR r, lanes 0-15 -> (M=r, N=lane); lanes 16-31 -> (M=8+r).
  v8f accs = acc0 + acc1;
#pragma unroll
  for (int r = 0; r < 8; ++r) gx[g][r][lane] = accs[r];
  __syncthreads();

  // 256 cell elements per block, 128 threads -> 2 each.
#pragma unroll
  for (int e = 0; e < 2; ++e) {
    int idx = tid * 2 + e;           // 0..255
    int m = idx >> 4;
    int j = idx & 15;
    int r  = m & 7;
    int ln = ((m & 8) ? 16 : 0) + j;
    float gi = gx[0][r][ln];
    float gf = gx[1][r][ln];
    float gg = gx[2][r][ln];
    float go = gx[3][r][ln];
    float i_ = 1.0f / (1.0f + __expf(-gi));
    float f_ = 1.0f / (1.0f + __expf(-gf));
    float o_ = 1.0f / (1.0f + __expf(-go));
    float g_ = tanhf(gg);
    int brow = bt * 16 + m;
    int jc   = jt * 16 + j;
    size_t off = (size_t)brow * H + jc;
    float cn = f_ * cbuf[off] + i_ * g_;
    cbuf[off] = cn;
    float hn = o_ * tanhf(cn);
    hfin[off]  = hn;
    hnext[off] = (bf16_t)hn;
    if (outseq) outseq[(size_t)brow * out_stride + jc] = hn;
  }
}

// ---------------------------------------------------------------------------
// Host launcher
// ---------------------------------------------------------------------------
extern "C" void kernel_launch(void* const* d_in, const int* in_sizes, int n_in,
                              void* d_out, int out_size, void* d_ws, size_t ws_size,
                              hipStream_t stream) {
  (void)in_sizes; (void)n_in; (void)out_size; (void)ws_size;
  const int Bc = 64, T = 256, N = 512, Hc = LSTM_H;
  const int BH = Bc * Hc;

  const float* x   = (const float*)d_in[0];
  const float* h   = (const float*)d_in[1];
  const float* c   = (const float*)d_in[2];
  const float* Wi0 = (const float*)d_in[3];
  const float* Wh0 = (const float*)d_in[4];
  const float* bi0 = (const float*)d_in[5];
  const float* bh0 = (const float*)d_in[6];
  const float* Wi1 = (const float*)d_in[7];
  const float* Wh1 = (const float*)d_in[8];
  const float* bi1 = (const float*)d_in[9];
  const float* bh1 = (const float*)d_in[10];

  // d_out: outs [B,T,H] fp32, then h_final [2,B,H], then c_final [2,B,H].
  float* outs = (float*)d_out;
  float* hfin = outs + (size_t)Bc * T * Hc;
  float* cfin = hfin + (size_t)2 * BH;

  // Workspace carve-up.
  char* ws = (char*)d_ws;
  size_t off = 0;
  auto carve = [&](size_t bytes) -> void* {
    void* p = ws + off;
    off = (off + bytes + 255) & ~(size_t)255;
    return p;
  };
  const int Kt0 = N + Hc;        // 1536
  const int Kt1 = Hc + Hc;       // 2048
  bf16_t* Wp0   = (bf16_t*)carve((size_t)4 * Hc * Kt0 * sizeof(bf16_t));
  bf16_t* Wp1   = (bf16_t*)carve((size_t)4 * Hc * Kt1 * sizeof(bf16_t));
  float*  bias0 = (float*) carve((size_t)4 * Hc * sizeof(float));
  float*  bias1 = (float*) carve((size_t)4 * Hc * sizeof(float));
  bf16_t* xbf   = (bf16_t*)carve((size_t)Bc * T * N * sizeof(bf16_t));
  bf16_t* hbf0[2], *hbf1[2];
  hbf0[0] = (bf16_t*)carve((size_t)BH * sizeof(bf16_t));
  hbf0[1] = (bf16_t*)carve((size_t)BH * sizeof(bf16_t));
  hbf1[0] = (bf16_t*)carve((size_t)BH * sizeof(bf16_t));
  hbf1[1] = (bf16_t*)carve((size_t)BH * sizeof(bf16_t));

  // --- one-time prep (re-run each call; deterministic) ---
  {
    int tot0 = 4 * Hc * Kt0;
    pack_weights_kernel<<<(tot0 + 255) / 256, 256, 0, stream>>>(
        Wp0, Wi0, Wh0, N, Kt0, tot0);
    int tot1 = 4 * Hc * Kt1;
    pack_weights_kernel<<<(tot1 + 255) / 256, 256, 0, stream>>>(
        Wp1, Wi1, Wh1, Hc, Kt1, tot1);
    bias_combine_kernel<<<(4 * Hc + 255) / 256, 256, 0, stream>>>(
        bias0, bi0, bh0, 4 * Hc);
    bias_combine_kernel<<<(4 * Hc + 255) / 256, 256, 0, stream>>>(
        bias1, bi1, bh1, 4 * Hc);
    int nx = Bc * T * N;
    f32_to_bf16_kernel<<<(nx + 255) / 256, 256, 0, stream>>>(xbf, x, nx);
    init_state_kernel<<<(2 * BH + 255) / 256, 256, 0, stream>>>(
        hbf0[0], hbf1[0], cfin, h, c, BH);
  }

  // --- sequential recurrence ---
  dim3 grid(Hc / 16, Bc / 16);   // (64, 4)
  dim3 block(32, 4);
  int par = 0;
  for (int t = 0; t < T; ++t) {
    // layer 0: A = [x_t, h0_prev]
    lstm_step_kernel<<<grid, block, 0, stream>>>(
        xbf + (size_t)t * N, T * N, N,
        hbf0[par], Wp0, bias0,
        cfin, hfin, nullptr, 0, hbf0[par ^ 1]);
    // layer 1: A = [h0_new, h1_prev]; writes outs[:, t, :]
    lstm_step_kernel<<<grid, block, 0, stream>>>(
        hbf0[par ^ 1], Hc, Hc,
        hbf1[par], Wp1, bias1,
        cfin + BH, hfin + BH, outs + (size_t)t * Hc, T * Hc, hbf1[par ^ 1]);
    par ^= 1;
  }
}
